// VectorQuantizer_51531017617467
// MI455X (gfx1250) — matrix operand order repaired and checked by reading, not compiled, not run on hardware
//
#include <hip/hip_runtime.h>
#include <math.h>

typedef __bf16 v16bf __attribute__((ext_vector_type(16)));
typedef float  v8f   __attribute__((ext_vector_type(8)));
typedef unsigned int u32x4 __attribute__((ext_vector_type(4)));
typedef int i32x4 __attribute__((ext_vector_type(4)));
typedef int i32x8 __attribute__((ext_vector_type(8)));

#define N_E    16384
#define BATCH  4096
#define DHALF  512
#define DFULL  1024
#define TOPK   10
#define BETA   0.25f

union ABReg { v16bf v; uint4 u[2]; };

static __device__ __forceinline__ unsigned short f2bf(float f) {
  unsigned u = __float_as_uint(f);
  u += 0x7FFFu + ((u >> 16) & 1u);   // round-to-nearest-even
  return (unsigned short)(u >> 16);
}
static __device__ __forceinline__ float bf2f(unsigned short h) {
  return __uint_as_float(((unsigned)h) << 16);
}
static __device__ __forceinline__ void wait_ds0() {
  asm volatile("s_wait_dscnt 0" ::: "memory");
}

// ---- TDM: async-load one 64-row x 64-elem bf16 tile (row stride Kd) to LDS,
// padding 16B per 128B row so LDS B-fragment reads are bank-conflict-free ----
static __device__ __forceinline__ void tdm_issue_tile(unsigned ldsOff,
                                                      const unsigned short* gptr,
                                                      int Kd) {
  const unsigned long long ga = (unsigned long long)(size_t)gptr;
  u32x4 g0;
  g0[0] = 1u;                                                  // count=1 (user D#)
  g0[1] = ldsOff;                                              // lds_addr (bytes)
  g0[2] = (unsigned)ga;                                        // global_addr[31:0]
  g0[3] = (unsigned)((ga >> 32) & 0x1FFFFFFull) | (2u << 30);  // addr[56:32] | type=2
  i32x8 g1;
  g1[0] = (1 << 16) | (1 << 20) | (4 << 22) | (3 << 25);
          // data_size=2B | pad_enable | pad_interval=32dw(128B) | pad_amount=4dw(16B)
  g1[1] = (Kd & 0xFFFF) << 16;       // tensor_dim0 = Kd (elements per codebook row)
  g1[2] = (N_E & 0xFFFF) << 16;      // tensor_dim0 hi=0 | tensor_dim1 = N_E rows
  g1[3] = 64 << 16;                  // tensor_dim1 hi=0 | tile_dim0 = 64 elems
  g1[4] = 64;                        // tile_dim1 = 64 rows | tile_dim2 = 0
  g1[5] = Kd;                        // tensor_dim0_stride low 32 (elements)
  g1[6] = 0;
  g1[7] = 0;
  i32x4 gz = {0, 0, 0, 0};
#if __clang_major__ >= 23
  i32x8 gz8 = {0, 0, 0, 0, 0, 0, 0, 0};
  __builtin_amdgcn_tensor_load_to_lds(g0, g1, gz, gz, gz8, 0);
#else
  __builtin_amdgcn_tensor_load_to_lds(g0, g1, gz, gz, 0);
#endif
}

// ---- zero scratch that must be deterministic across graph replays ----
__global__ void k_zero(unsigned* __restrict__ flags, float* __restrict__ lossSums) {
  const int i = blockIdx.x * 256 + threadIdx.x;
  if (i < 3 * N_E) flags[i] = 0u;
  if (i < 3) lossSums[i] = 0.0f;
}

// ---- L2-normalize z rows (two independent 512-d halves) -> bf16 ----
__global__ void k_norm_z(const float* __restrict__ z, unsigned short* __restrict__ zn) {
  const int row = blockIdx.x;         // 4096
  const int t = threadIdx.x;          // 256 threads, 4 floats each
  const float4 v = ((const float4*)(z + (size_t)row * DFULL))[t];
  const int half = (t >= 128) ? 1 : 0;
  float s = v.x * v.x + v.y * v.y + v.z * v.z + v.w * v.w;
  __shared__ float acc[2];
  if (t < 2) acc[t] = 0.0f;
  __syncthreads();
  atomicAdd(&acc[half], s);
  __syncthreads();
  const float inv = rsqrtf(acc[half] + 1e-12f);
  ushort4 o;
  o.x = f2bf(v.x * inv); o.y = f2bf(v.y * inv);
  o.z = f2bf(v.z * inv); o.w = f2bf(v.w * inv);
  ((ushort4*)(zn + (size_t)row * DFULL))[t] = o;
}

// ---- L2-normalize one 512-d codebook row -> bf16 at (dstStride,colOff) ----
__global__ void k_norm_w(const float* __restrict__ w, unsigned short* __restrict__ dst,
                         int dstStride, int dstColOff) {
  const int row = blockIdx.x;         // 16384
  const int t = threadIdx.x;          // 128 threads, 4 floats each
  const float4 v = ((const float4*)(w + (size_t)row * DHALF))[t];
  float s = v.x * v.x + v.y * v.y + v.z * v.z + v.w * v.w;
  __shared__ float acc;
  if (t == 0) acc = 0.0f;
  __syncthreads();
  atomicAdd(&acc, s);
  __syncthreads();
  const float inv = rsqrtf(acc + 1e-12f);
  ushort4 o;
  o.x = f2bf(v.x * inv); o.y = f2bf(v.y * inv);
  o.z = f2bf(v.z * inv); o.w = f2bf(v.w * inv);
  ((ushort4*)(dst + (size_t)row * dstStride + dstColOff))[t] = o;
}

// ---- fused: TDM-staged bf16 WMMA distance GEMM + top-10 + softmax gather ----
// Block = 128 threads = 4 waves. Each wave owns 32 rows (2 WMMA M-tiles) and
// scans all N_E codebook rows in 64-wide sub-tiles (4 N-tiles, 8 accumulators).
// B tiles (64 rows x 64 elems) are double-buffered in LDS via tensor_load_to_lds.
__global__ void __launch_bounds__(128)
k_vq(const unsigned short* __restrict__ zn, int zColOff,
     const unsigned short* __restrict__ wn,
     const float* __restrict__ z,
     float* __restrict__ outZq,
     unsigned* __restrict__ flags,
     float* __restrict__ lossSum,
     int Kd, float distConst)
{
  __shared__ __align__(16) unsigned short ldsB[2][64 * 72]; // 72 = 64 + 8 pad (TDM pad)
  __shared__ float ldsD[4][32][65];
  __shared__ float ldsW[4][32][TOPK];
  __shared__ int   ldsI[4][32][TOPK];

  const int tid  = threadIdx.x;
  const int wave = tid >> 5;
  const int lane = tid & 31;
  const int m16  = lane & 15;
  const int grp  = lane >> 4;
  const int rowBase = blockIdx.x * 128 + wave * 32;

  float tv[TOPK]; int ti[TOPK];
#pragma unroll
  for (int q = 0; q < TOPK; ++q) { tv[q] = 3.0e38f; ti[q] = 0; }

  // A fragment (contiguous k order, same as B): lane m16 holds K chunk
  // [grp*16 .. +15] of its row.
  const unsigned short* Abase0 =
      zn + (size_t)(rowBase + m16) * DFULL + zColOff + grp * 16;
  const unsigned short* Abase1 = Abase0 + (size_t)16 * DFULL;

  const unsigned bufOff0 = (unsigned)(size_t)&ldsB[0][0];
  const unsigned bufOff1 = (unsigned)(size_t)&ldsB[1][0];

  if (wave == 0) tdm_issue_tile(bufOff0, wn, Kd);   // prologue: first tile
  int cur = 0;

  for (int nb = 0; nb < N_E; nb += 64) {
    v8f accv[2][4] = {};
    for (int kc = 0; kc < Kd; kc += 64) {
      if (wave == 0) __builtin_amdgcn_s_wait_tensorcnt(0);  // current tile landed
      __syncthreads();                                      // visible to all waves
      int nnb = nb, nkc = kc + 64;
      if (nkc == Kd) { nkc = 0; nnb += 64; }
      if (wave == 0 && nnb < N_E)                           // overlap next tile
        tdm_issue_tile(cur ? bufOff0 : bufOff1, wn + (size_t)nnb * Kd + nkc, Kd);

      const unsigned short* Bl = &ldsB[cur][0];
#pragma unroll
      for (int s = 0; s < 2; ++s) {
        const int kk = s * 32;
        // issue ALL fragment loads first (A via VMEM, 4 B tiles via DS) so the
        // scheduler can retire them with partial waits instead of dscnt==0 stalls
        ABReg a0, a1, b0, b1, b2, b3;
        a0.u[0] = *(const uint4*)(Abase0 + kc + kk);
        a0.u[1] = *(const uint4*)(Abase0 + kc + kk + 8);
        a1.u[0] = *(const uint4*)(Abase1 + kc + kk);
        a1.u[1] = *(const uint4*)(Abase1 + kc + kk + 8);
        const unsigned short* bp = Bl + m16 * 72 + kk + grp * 16;
        b0.u[0] = *(const uint4*)(bp);
        b0.u[1] = *(const uint4*)(bp + 8);
        b1.u[0] = *(const uint4*)(bp + 16 * 72);
        b1.u[1] = *(const uint4*)(bp + 16 * 72 + 8);
        b2.u[0] = *(const uint4*)(bp + 32 * 72);
        b2.u[1] = *(const uint4*)(bp + 32 * 72 + 8);
        b3.u[0] = *(const uint4*)(bp + 48 * 72);
        b3.u[1] = *(const uint4*)(bp + 48 * 72 + 8);
        accv[0][0] = __builtin_amdgcn_wmma_f32_16x16x32_bf16(
            false, a0.v, false, b0.v, (short)0, accv[0][0], false, false);
        accv[1][0] = __builtin_amdgcn_wmma_f32_16x16x32_bf16(
            false, a1.v, false, b0.v, (short)0, accv[1][0], false, false);
        accv[0][1] = __builtin_amdgcn_wmma_f32_16x16x32_bf16(
            false, a0.v, false, b1.v, (short)0, accv[0][1], false, false);
        accv[1][1] = __builtin_amdgcn_wmma_f32_16x16x32_bf16(
            false, a1.v, false, b1.v, (short)0, accv[1][1], false, false);
        accv[0][2] = __builtin_amdgcn_wmma_f32_16x16x32_bf16(
            false, a0.v, false, b2.v, (short)0, accv[0][2], false, false);
        accv[1][2] = __builtin_amdgcn_wmma_f32_16x16x32_bf16(
            false, a1.v, false, b2.v, (short)0, accv[1][2], false, false);
        accv[0][3] = __builtin_amdgcn_wmma_f32_16x16x32_bf16(
            false, a0.v, false, b3.v, (short)0, accv[0][3], false, false);
        accv[1][3] = __builtin_amdgcn_wmma_f32_16x16x32_bf16(
            false, a1.v, false, b3.v, (short)0, accv[1][3], false, false);
      }
      cur ^= 1;
    }

    // distances -> LDS (C layout: vgpr r, lanes0-15 M=r, lanes16-31 M=r+8)
#pragma unroll
    for (int mt = 0; mt < 2; ++mt)
#pragma unroll
      for (int j = 0; j < 4; ++j)
#pragma unroll
        for (int r = 0; r < 8; ++r)
          ldsD[wave][mt * 16 + r + 8 * grp][j * 16 + m16] =
              distConst - 2.0f * accv[mt][j][r];
    wait_ds0();

    // every lane owns one of the wave's 32 rows: running sorted top-10
    for (int n = 0; n < 64; ++n) {
      float d = ldsD[wave][lane][n];
      if (d < tv[TOPK - 1]) {
        tv[TOPK - 1] = d; ti[TOPK - 1] = nb + n;
#pragma unroll
        for (int q = TOPK - 1; q > 0; --q) {
          if (tv[q] < tv[q - 1]) {
            float tf = tv[q]; tv[q] = tv[q - 1]; tv[q - 1] = tf;
            int tt = ti[q]; ti[q] = ti[q - 1]; ti[q - 1] = tt;
          }
        }
      }
    }
    wait_ds0();
  }

  // softmax(-d) over the 10 best; publish weights/indices; mark usage
  {
    float mx = tv[0], s = 0.0f, e[TOPK];
#pragma unroll
    for (int q = 0; q < TOPK; ++q) { e[q] = __expf(mx - tv[q]); s += e[q]; }
    const float invs = 1.0f / s;
#pragma unroll
    for (int q = 0; q < TOPK; ++q) {
      ldsW[wave][lane][q] = e[q] * invs;
      ldsI[wave][lane][q] = ti[q];
      flags[ti[q]] = 1u;              // same-value stores: race-free
    }
  }
  wait_ds0();

  // gather: z_q = sum_k w_k * emb_n[idx_k]; loss partial vs raw z slice
  float lossLocal = 0.0f;
  for (int m = 0; m < 32; ++m) {
    const int row = rowBase + m;
    float w[TOPK]; int id[TOPK];
#pragma unroll
    for (int q = 0; q < TOPK; ++q) { w[q] = ldsW[wave][m][q]; id[q] = ldsI[wave][m][q]; }
    for (int c = lane; c < Kd; c += 32) {
      float sum = 0.0f;
#pragma unroll
      for (int q = 0; q < TOPK; ++q)
        sum += w[q] * bf2f(wn[(size_t)id[q] * Kd + c]);
      const float zr = z[(size_t)row * DFULL + zColOff + c];
      const float df = sum - zr;
      lossLocal += df * df;
      outZq[(size_t)row * Kd + c] = sum;   // straight-through fwd value == z_q
    }
  }
  atomicAdd(lossSum, lossLocal);
}

// ---- split zq_s into its text/graph halves ----
__global__ void k_split(const float* __restrict__ zqs,
                        float* __restrict__ st, float* __restrict__ sg) {
  const size_t i = (size_t)blockIdx.x * blockDim.x + threadIdx.x;  // 4096*512
  const size_t row = i / DHALF, c = i - row * DHALF;
  st[i] = zqs[row * DFULL + c];
  sg[i] = zqs[row * DFULL + DHALF + c];
}

// ---- scalars: vq, commit (= beta*vq in fwd value), usage ----
__global__ void k_final(const unsigned* __restrict__ flags,
                        const float* __restrict__ lossSums,
                        float* __restrict__ sc) {
  const int p = blockIdx.x;           // 0=shared,1=text,2=graph
  __shared__ unsigned cnt;
  if (threadIdx.x == 0) cnt = 0;
  __syncthreads();
  unsigned local = 0;
  for (int i = threadIdx.x; i < N_E; i += blockDim.x)
    local += (flags[p * N_E + i] != 0u);
  atomicAdd(&cnt, local);
  __syncthreads();
  if (threadIdx.x == 0) {
    const float denom = (p == 0) ? (float)(BATCH) * (float)DFULL
                                 : (float)(BATCH) * (float)DHALF;
    const float vq = lossSums[p] / denom;
    sc[2 * p]     = vq;
    sc[2 * p + 1] = BETA * vq;
    sc[6 + p]     = (float)cnt / (float)N_E;
  }
}

extern "C" void kernel_launch(void* const* d_in, const int* in_sizes, int n_in,
                              void* d_out, int out_size, void* d_ws, size_t ws_size,
                              hipStream_t stream) {
  (void)in_sizes; (void)n_in; (void)out_size; (void)ws_size;
  const float* z   = (const float*)d_in[0];
  const float* wt  = (const float*)d_in[1];
  const float* wg  = (const float*)d_in[2];
  const float* wst = (const float*)d_in[3];
  const float* wsg = (const float*)d_in[4];
  float* out = (float*)d_out;

  // workspace layout (~72 MB)
  unsigned short* zn  = (unsigned short*)d_ws;              // 4096 x 1024 bf16
  unsigned short* wsn = zn  + (size_t)BATCH * DFULL;        // 16384 x 1024 bf16 (shared concat)
  unsigned short* wtn = wsn + (size_t)N_E * DFULL;          // 16384 x 512 bf16
  unsigned short* wgn = wtn + (size_t)N_E * DHALF;          // 16384 x 512 bf16
  unsigned* flags     = (unsigned*)(wgn + (size_t)N_E * DHALF); // 3 x 16384
  float* lossSums     = (float*)(flags + 3 * N_E);          // 3

  const size_t OFF_ST  = (size_t)BATCH * DFULL;
  const size_t OFF_SG  = OFF_ST  + (size_t)BATCH * DHALF;
  const size_t OFF_ZQT = OFF_SG  + (size_t)BATCH * DHALF;
  const size_t OFF_ZQG = OFF_ZQT + (size_t)BATCH * DHALF;
  const size_t OFF_SC  = OFF_ZQG + (size_t)BATCH * DHALF;   // 9 scalars

  k_zero<<<(3 * N_E + 255) / 256, 256, 0, stream>>>(flags, lossSums);
  k_norm_z<<<BATCH, 256, 0, stream>>>(z, zn);
  k_norm_w<<<N_E, 128, 0, stream>>>(wst, wsn, DFULL, 0);
  k_norm_w<<<N_E, 128, 0, stream>>>(wsg, wsn, DFULL, DHALF);
  k_norm_w<<<N_E, 128, 0, stream>>>(wt,  wtn, DHALF, 0);
  k_norm_w<<<N_E, 128, 0, stream>>>(wg,  wgn, DHALF, 0);

  // shared: K=1024, ||.||^2 const = 4;  text/graph: K=512, const = 2
  k_vq<<<BATCH / 128, 128, 0, stream>>>(zn, 0,     wsn, z, out,           flags,           lossSums + 0, DFULL, 4.0f);
  k_vq<<<BATCH / 128, 128, 0, stream>>>(zn, 0,     wtn, z, out + OFF_ZQT, flags + N_E,     lossSums + 1, DHALF, 2.0f);
  k_vq<<<BATCH / 128, 128, 0, stream>>>(zn, DHALF, wgn, z, out + OFF_ZQG, flags + 2 * N_E, lossSums + 2, DHALF, 2.0f);

  k_split<<<(BATCH * DHALF + 255) / 256, 256, 0, stream>>>(out, out + OFF_ST, out + OFF_SG);
  k_final<<<3, 256, 0, stream>>>(flags, lossSums, out + OFF_SC);
}
